// DenseCL_70935679861121
// MI455X (gfx1250) — matrix operand order repaired and checked by reading, not compiled, not run on hardware
//
#include <hip/hip_runtime.h>
#include <hip/hip_bf16.h>

// ---------------------------------------------------------------------------
// DenseCL forward loss on MI455X (gfx1250, wave32, WMMA + async-LDS).
// Heavy GEMMs: bf16x3 split-float emulation of fp32 on v_wmma_f32_16x16x32_bf16
// (compute-bound: ~126 GFLOP vs ~200MB -> AI ~600 FLOP/B >> HBM roofline knee).
// Data movement: cooperative GLOBAL_LOAD_ASYNC_TO_LDS_B128 (ASYNCcnt) into a
// double-buffered 64KB LDS stage; fragments then read via ds_load_b128 in the
// WMMA layouts (A and B use the same k order):
//   A 16x32 bf16 : lane l holds row m=l&15; ks  k0+16h .. k0+16h+15 (contig), h=l>>4
//   B 32x16 bf16 : lane l holds col n=l&15; ks  k0+16h .. k0+16h+15 (contig)
//   C 16x16 f32  : vgpr v, lane l -> (m = v + 8*(l>>4), n = l&15)
// Weights pre-transposed to [N,K] so B chunks are contiguous in memory.
// ---------------------------------------------------------------------------

typedef __attribute__((ext_vector_type(16))) __bf16 bf16x16;
typedef __attribute__((ext_vector_type(8)))  __bf16 bf16x8;
typedef __attribute__((ext_vector_type(8)))  float  f32x8;

#define TAU_INV 5.0f          // 1/0.2
#define L2_EPS  1e-12f

__device__ __forceinline__ void split_bf16(float x, __bf16& h, __bf16& l) {
  __bf16 hh = (__bf16)x;          // RNE convert
  h = hh;
  l = (__bf16)(x - (float)hh);    // residual
}

// async DMA: 16B per lane, global -> LDS, no VGPR data; tracked by ASYNCcnt.
// LDS destination address taken from a VGPR (low 32 bits of flat shared pointer == LDS offset).
__device__ __forceinline__ void async_copy_b128(const void* gsrc, void* ldst) {
  unsigned ldsOff = (unsigned)(size_t)ldst;
  asm volatile("global_load_async_to_lds_b128 %0, %1, off"
               :: "v"(ldsOff), "v"(gsrc) : "memory");
}
#define ASYNC_WAIT(n) asm volatile("s_wait_asynccnt " #n ::: "memory")

// ---------------------------- pre-pass kernels -----------------------------

// elementwise fp32 -> (bf16 hi, bf16 lo)
__global__ void split_kernel(const float* __restrict__ X,
                             __bf16* __restrict__ H, __bf16* __restrict__ L, int n) {
  int i = blockIdx.x * blockDim.x + threadIdx.x;
  if (i < n) { __bf16 h, l; split_bf16(X[i], h, l); H[i] = h; L[i] = l; }
}

// global average pool over P pixels, then split: F[B,P,C] -> G[B,C] (hi/lo)
__global__ void pool_split_kernel(const float* __restrict__ F,
                                  __bf16* __restrict__ Gh, __bf16* __restrict__ Gl,
                                  int B, int P, int C) {
  int i = blockIdx.x * blockDim.x + threadIdx.x;
  if (i >= B * C) return;
  int b = i / C, c = i - b * C;
  const float* p = F + (size_t)b * P * C + c;
  float s = 0.f;
  for (int j = 0; j < P; ++j) s += p[(size_t)j * C];
  s *= (1.0f / 49.0f);
  __bf16 h, l; split_bf16(s, h, l);
  Gh[i] = h; Gl[i] = l;
}

// W[K,N] (row major) -> T[N,K] split to bf16 hi/lo, 32x32 LDS tiles
__global__ __launch_bounds__(1024)
void transpose_split_kernel(const float* __restrict__ W,
                            __bf16* __restrict__ Th, __bf16* __restrict__ Tl,
                            int K, int N) {
  __shared__ float t[32][33];
  int k0 = blockIdx.y * 32, n0 = blockIdx.x * 32;
  int tx = threadIdx.x, ty = threadIdx.y;
  int k = k0 + ty, n = n0 + tx;
  if (k < K && n < N) t[ty][tx] = W[(size_t)k * N + n];
  __syncthreads();
  int nn = n0 + ty, kk = k0 + tx;
  if (nn < N && kk < K) {
    __bf16 h, l; split_bf16(t[tx][ty], h, l);
    Th[(size_t)nn * K + kk] = h;
    Tl[(size_t)nn * K + kk] = l;
  }
}

// ------------------------------ WMMA GEMM ----------------------------------
// C[M,N] = act(A[M,K] @ Bt[N,K]^T + bias), A/Bt given as bf16 hi/lo pairs.
// MODE 0: ReLU, output split to bf16 hi/lo (layer-1).  MODE 1: fp32 out.
// Block = 256 thr (8 waves, 2x4), block tile 128x128, wave tile 64x32.
// Double-buffered async global->LDS staging, 24 WMMAs per k-step per wave.
template<int MODE>
__global__ __launch_bounds__(256)
void gemm_bf16x3(const __bf16* __restrict__ Ah, const __bf16* __restrict__ Al,
                 const __bf16* __restrict__ Bth, const __bf16* __restrict__ Btl,
                 const float* __restrict__ bias,
                 int M, int N, int K,
                 __bf16* __restrict__ Oh, __bf16* __restrict__ Ol,
                 float* __restrict__ Of) {
  // LDS stage: [2 buffers][tile], A: 128 rows x 32 k, B: 128 cols x 32 k
  __shared__ __bf16 sAh[2][128 * 32];
  __shared__ __bf16 sAl[2][128 * 32];
  __shared__ __bf16 sBh[2][128 * 32];
  __shared__ __bf16 sBl[2][128 * 32];

  const int tid  = threadIdx.x;
  const int lane = tid & 31;
  const int w    = tid >> 5;
  const int wm = w & 1, wn = w >> 1;                 // 2 (m) x 4 (n) wave grid
  const int rowBase0 = blockIdx.x * 128;
  const int colBase0 = blockIdx.y * 128;
  const int half = lane >> 4;
  const int l15  = lane & 15;

  // cooperative copy assignment: 512 16B-chunks per tile, 2 per thread.
  // chunk c: row = c>>2 (64B per 32-elem row), sub = c&3 (8 elems).
  const int c0 = tid, c1 = tid + 256;
  const int r0 = c0 >> 2, s0 = (c0 & 3) * 8;
  const int r1 = c1 >> 2, s1 = (c1 & 3) * 8;

  auto issue = [&](int buf, int k0) {
    // A hi/lo
    async_copy_b128(Ah  + (size_t)(rowBase0 + r0) * K + k0 + s0, &sAh[buf][r0 * 32 + s0]);
    async_copy_b128(Ah  + (size_t)(rowBase0 + r1) * K + k0 + s1, &sAh[buf][r1 * 32 + s1]);
    async_copy_b128(Al  + (size_t)(rowBase0 + r0) * K + k0 + s0, &sAl[buf][r0 * 32 + s0]);
    async_copy_b128(Al  + (size_t)(rowBase0 + r1) * K + k0 + s1, &sAl[buf][r1 * 32 + s1]);
    // B hi/lo (rows are output-cols of Bt[N,K])
    async_copy_b128(Bth + (size_t)(colBase0 + r0) * K + k0 + s0, &sBh[buf][r0 * 32 + s0]);
    async_copy_b128(Bth + (size_t)(colBase0 + r1) * K + k0 + s1, &sBh[buf][r1 * 32 + s1]);
    async_copy_b128(Btl + (size_t)(colBase0 + r0) * K + k0 + s0, &sBl[buf][r0 * 32 + s0]);
    async_copy_b128(Btl + (size_t)(colBase0 + r1) * K + k0 + s1, &sBl[buf][r1 * 32 + s1]);
  };

  f32x8 acc[4][2] = {};

  issue(0, 0);                                       // prologue: fill buf0 (+8)
  for (int k0 = 0; k0 < K; k0 += 32) {
    const int cur = (k0 >> 5) & 1;
    const bool more = (k0 + 32) < K;
    if (more) {
      issue(cur ^ 1, k0 + 32);                       // prefetch next (+8)
      ASYNC_WAIT(8);                                 // in-order: cur's 8 done
    } else {
      ASYNC_WAIT(0);                                 // drain
    }
    __syncthreads();                                 // cur tile visible to all

    bf16x16 bh[2], bl[2];
#pragma unroll
    for (int ni = 0; ni < 2; ++ni) {
      int nl = wn * 32 + ni * 16 + l15;
      bh[ni] = *(const bf16x16*)&sBh[cur][nl * 32 + 16 * half];
      bl[ni] = *(const bf16x16*)&sBl[cur][nl * 32 + 16 * half];
    }
#pragma unroll
    for (int mi = 0; mi < 4; ++mi) {
      int ml = wm * 64 + mi * 16 + l15;
      bf16x8 a0 = *(const bf16x8*)&sAh[cur][ml * 32 + 16 * half];
      bf16x8 a1 = *(const bf16x8*)&sAh[cur][ml * 32 + 16 * half + 8];
      bf16x8 a2 = *(const bf16x8*)&sAl[cur][ml * 32 + 16 * half];
      bf16x8 a3 = *(const bf16x8*)&sAl[cur][ml * 32 + 16 * half + 8];
      bf16x16 ah, al;
#pragma unroll
      for (int e = 0; e < 8; ++e) {
        ah[e] = a0[e]; ah[e + 8] = a1[e];
        al[e] = a2[e]; al[e + 8] = a3[e];
      }
#pragma unroll
      for (int ni = 0; ni < 2; ++ni) {
        acc[mi][ni] = __builtin_amdgcn_wmma_f32_16x16x32_bf16(
            false, ah, false, bh[ni], (short)0, acc[mi][ni], false, false);
        acc[mi][ni] = __builtin_amdgcn_wmma_f32_16x16x32_bf16(
            false, ah, false, bl[ni], (short)0, acc[mi][ni], false, false);
        acc[mi][ni] = __builtin_amdgcn_wmma_f32_16x16x32_bf16(
            false, al, false, bh[ni], (short)0, acc[mi][ni], false, false);
      }
    }
    __syncthreads();                                 // done reading cur buffer
  }

  const int rowBase = rowBase0 + wm * 64;
  const int colBase = colBase0 + wn * 32;
#pragma unroll
  for (int mi = 0; mi < 4; ++mi) {
#pragma unroll
    for (int ni = 0; ni < 2; ++ni) {
      int n = colBase + ni * 16 + l15;
      float bn = bias[n];
#pragma unroll
      for (int v = 0; v < 8; ++v) {
        int m = rowBase + mi * 16 + v + 8 * half;
        float x = acc[mi][ni][v] + bn;
        if (MODE == 0) {
          x = x > 0.f ? x : 0.f;
          __bf16 h, l; split_bf16(x, h, l);
          Oh[(size_t)m * N + n] = h;
          Ol[(size_t)m * N + n] = l;
        } else {
          Of[(size_t)m * N + n] = x;
        }
      }
    }
  }
}

// ------------------------------ tail kernels -------------------------------

// l2-normalize rows in place; one wave32 per row
__global__ void l2norm_kernel(float* __restrict__ X, int M, int D) {
  int row  = blockIdx.x * (blockDim.x >> 5) + (threadIdx.x >> 5);
  int lane = threadIdx.x & 31;
  if (row >= M) return;
  float* p = X + (size_t)row * D;
  float ss = 0.f;
  for (int j = lane; j < D; j += 32) { float v = p[j]; ss += v * v; }
  for (int off = 16; off > 0; off >>= 1) ss += __shfl_xor(ss, off, 32);
  float r = rsqrtf(fmaxf(ss, L2_EPS));
  for (int j = lane; j < D; j += 32) p[j] *= r;
}

// global InfoNCE partials: one block (128 thr) per row, N=D=128
__global__ __launch_bounds__(128)
void global_nce_kernel(const float* __restrict__ Q, const float* __restrict__ Km,
                       float* __restrict__ partial) {
  __shared__ float q[128];
  __shared__ float red[128];
  int i = blockIdx.x, t = threadIdx.x;
  q[t] = Q[(size_t)i * 128 + t];
  __syncthreads();
  const float* kp = Km + (size_t)t * 128;
  float dot = 0.f;
  for (int c = 0; c < 128; ++c) dot += q[c] * kp[c];
  float logit = dot * TAU_INV;
  red[t] = logit; __syncthreads();
  for (int s = 64; s > 0; s >>= 1) { if (t < s) red[t] = fmaxf(red[t], red[t + s]); __syncthreads(); }
  float mx = red[0]; __syncthreads();
  red[t] = expf(logit - mx); __syncthreads();
  for (int s = 64; s > 0; s >>= 1) { if (t < s) red[t] += red[t + s]; __syncthreads(); }
  float lse = mx + logf(red[0]);
  if (t == i) partial[i] = lse - logit;   // diagonal = own logit
}

// per-batch 49x49 similarity argmax -> global matched row index
__global__ __launch_bounds__(64)
void argmax_kernel(const float* __restrict__ Qd, const float* __restrict__ Kd,
                   int* __restrict__ midx) {
  __shared__ float qs[49 * 128];
  __shared__ float ks[49 * 128];
  int b = blockIdx.x, t = threadIdx.x;
  const float* qb = Qd + (size_t)b * 49 * 128;
  const float* kb = Kd + (size_t)b * 49 * 128;
  for (int j = t; j < 49 * 128; j += 64) { qs[j] = qb[j]; ks[j] = kb[j]; }
  __syncthreads();
  if (t < 49) {
    float best = -3.4e38f; int bi = 0;
    for (int m = 0; m < 49; ++m) {
      float d = 0.f;
      for (int c = 0; c < 128; ++c) d += qs[t * 128 + c] * ks[m * 128 + c];
      if (d > best) { best = d; bi = m; }   // strict >: first max, like argmax
    }
    midx[b * 49 + t] = b * 49 + bi;
  }
}

// dense InfoNCE partials with online logsumexp; one block (256 thr) per row
__global__ __launch_bounds__(256)
void dense_nce_kernel(const float* __restrict__ Qd, const float* __restrict__ Kd,
                      const int* __restrict__ midx, float* __restrict__ partial,
                      int Nrows) {
  __shared__ float q[128];
  __shared__ float rm[256], rs[256];
  int i = blockIdx.x, t = threadIdx.x;
  for (int j = t; j < 128; j += 256) q[j] = Qd[(size_t)i * 128 + j];
  __syncthreads();
  float mx = -3.4e38f, sum = 0.f;
  for (int j = t; j < Nrows; j += 256) {
    const float* kp = Kd + (size_t)midx[j] * 128;
    float d = 0.f;
    for (int c = 0; c < 128; ++c) d += q[c] * kp[c];
    d *= TAU_INV;
    if (d > mx) { sum = sum * expf(mx - d) + 1.f; mx = d; }
    else        { sum += expf(d - mx); }
  }
  rm[t] = mx; rs[t] = sum; __syncthreads();
  for (int s = 128; s > 0; s >>= 1) {
    if (t < s) {
      float m2 = rm[t + s], s2 = rs[t + s];
      if (m2 > rm[t]) { rs[t] = rs[t] * expf(rm[t] - m2) + s2; rm[t] = m2; }
      else            { rs[t] += s2 * expf(m2 - rm[t]); }
    }
    __syncthreads();
  }
  if (t == 0) {
    float lse = rm[0] + logf(rs[0]);
    const float* kp = Kd + (size_t)midx[i] * 128;
    float pos = 0.f;
    for (int c = 0; c < 128; ++c) pos += q[c] * kp[c];
    partial[i] = lse - pos * TAU_INV;
  }
}

__global__ __launch_bounds__(256)
void final_kernel(const float* __restrict__ pg, int ng,
                  const float* __restrict__ pd, int nd, float* __restrict__ out) {
  __shared__ float r[256];
  int t = threadIdx.x;
  float s = 0.f;
  for (int j = t; j < ng; j += 256) s += pg[j];
  r[t] = s; __syncthreads();
  for (int k = 128; k > 0; k >>= 1) { if (t < k) r[t] += r[t + k]; __syncthreads(); }
  float lg = r[0] / (float)ng; __syncthreads();
  s = 0.f;
  for (int j = t; j < nd; j += 256) s += pd[j];
  r[t] = s; __syncthreads();
  for (int k = 128; k > 0; k >>= 1) { if (t < k) r[t] += r[t + k]; __syncthreads(); }
  float ld = r[0] / (float)nd;
  if (t == 0) out[0] = 0.5f * lg + 0.5f * ld;   // LW = 0.5
}

// ------------------------------- launcher ----------------------------------

extern "C" void kernel_launch(void* const* d_in, const int* in_sizes, int n_in,
                              void* d_out, int out_size, void* d_ws, size_t ws_size,
                              hipStream_t stream) {
  constexpr int B = 128, P = 49, C = 2048, Dh = 2048, De = 128;
  constexpr int M = B * P;                       // 6272 dense rows

  const float* feat_q = (const float*)d_in[0];
  const float* feat_k = (const float*)d_in[1];
  const float* Wg1 = (const float*)d_in[2];  const float* bg1 = (const float*)d_in[3];
  const float* Wg2 = (const float*)d_in[4];  const float* bg2 = (const float*)d_in[5];
  const float* Wd1 = (const float*)d_in[6];  const float* bd1 = (const float*)d_in[7];
  const float* Wd2 = (const float*)d_in[8];  const float* bd2 = (const float*)d_in[9];
  const float* mWg1 = (const float*)d_in[10]; const float* mbg1 = (const float*)d_in[11];
  const float* mWg2 = (const float*)d_in[12]; const float* mbg2 = (const float*)d_in[13];
  const float* mWd1 = (const float*)d_in[14]; const float* mbd1 = (const float*)d_in[15];
  const float* mWd2 = (const float*)d_in[16]; const float* mbd2 = (const float*)d_in[17];
  float* out = (float*)d_out;

  // ---- bump allocator in workspace (~290 MB) ----
  char* ws = (char*)d_ws;
  size_t off = 0;
  auto alloc = [&](size_t bytes) -> void* {
    off = (off + 255) & ~(size_t)255;
    void* p = ws + off; off += bytes; return p;
  };
  auto bfbuf = [&](size_t elems) { return (__bf16*)alloc(elems * sizeof(__bf16)); };
  auto fbuf  = [&](size_t elems) { return (float*)alloc(elems * sizeof(float)); };

  __bf16 *XqH = bfbuf((size_t)M*C), *XqL = bfbuf((size_t)M*C);
  __bf16 *XkH = bfbuf((size_t)M*C), *XkL = bfbuf((size_t)M*C);
  __bf16 *GqH = bfbuf((size_t)B*C), *GqL = bfbuf((size_t)B*C);
  __bf16 *GkH = bfbuf((size_t)B*C), *GkL = bfbuf((size_t)B*C);
  __bf16 *Wd1tH = bfbuf((size_t)Dh*C), *Wd1tL = bfbuf((size_t)Dh*C);
  __bf16 *mWd1tH = bfbuf((size_t)Dh*C), *mWd1tL = bfbuf((size_t)Dh*C);
  __bf16 *Wg1tH = bfbuf((size_t)Dh*C), *Wg1tL = bfbuf((size_t)Dh*C);
  __bf16 *mWg1tH = bfbuf((size_t)Dh*C), *mWg1tL = bfbuf((size_t)Dh*C);
  __bf16 *Wd2tH = bfbuf((size_t)De*Dh), *Wd2tL = bfbuf((size_t)De*Dh);
  __bf16 *mWd2tH = bfbuf((size_t)De*Dh), *mWd2tL = bfbuf((size_t)De*Dh);
  __bf16 *Wg2tH = bfbuf((size_t)De*Dh), *Wg2tL = bfbuf((size_t)De*Dh);
  __bf16 *mWg2tH = bfbuf((size_t)De*Dh), *mWg2tL = bfbuf((size_t)De*Dh);
  __bf16 *HdqH = bfbuf((size_t)M*Dh), *HdqL = bfbuf((size_t)M*Dh);
  __bf16 *HdkH = bfbuf((size_t)M*Dh), *HdkL = bfbuf((size_t)M*Dh);
  __bf16 *HgqH = bfbuf((size_t)B*Dh), *HgqL = bfbuf((size_t)B*Dh);
  __bf16 *HgkH = bfbuf((size_t)B*Dh), *HgkL = bfbuf((size_t)B*Dh);
  float *qd = fbuf((size_t)M*De), *kd = fbuf((size_t)M*De);
  float *qg = fbuf((size_t)B*De), *kg = fbuf((size_t)B*De);
  int   *midx = (int*)alloc((size_t)M * sizeof(int));
  float *pd = fbuf(M), *pg = fbuf(B);

  // ---- 1) split activations, pool+split globals ----
  int nX = M * C;
  split_kernel<<<nX / 256, 256, 0, stream>>>(feat_q, XqH, XqL, nX);
  split_kernel<<<nX / 256, 256, 0, stream>>>(feat_k, XkH, XkL, nX);
  pool_split_kernel<<<(B * C) / 256, 256, 0, stream>>>(feat_q, GqH, GqL, B, P, C);
  pool_split_kernel<<<(B * C) / 256, 256, 0, stream>>>(feat_k, GkH, GkL, B, P, C);

  // ---- 2) transpose + split weights ----
  dim3 tb(32, 32);
  dim3 tg1(Dh / 32, C / 32);     // [C,Dh] -> [Dh,C]
  transpose_split_kernel<<<tg1, tb, 0, stream>>>(Wd1,  Wd1tH,  Wd1tL,  C, Dh);
  transpose_split_kernel<<<tg1, tb, 0, stream>>>(mWd1, mWd1tH, mWd1tL, C, Dh);
  transpose_split_kernel<<<tg1, tb, 0, stream>>>(Wg1,  Wg1tH,  Wg1tL,  C, Dh);
  transpose_split_kernel<<<tg1, tb, 0, stream>>>(mWg1, mWg1tH, mWg1tL, C, Dh);
  dim3 tg2(De / 32, Dh / 32);    // [Dh,De] -> [De,Dh]
  transpose_split_kernel<<<tg2, tb, 0, stream>>>(Wd2,  Wd2tH,  Wd2tL,  Dh, De);
  transpose_split_kernel<<<tg2, tb, 0, stream>>>(mWd2, mWd2tH, mWd2tL, Dh, De);
  transpose_split_kernel<<<tg2, tb, 0, stream>>>(Wg2,  Wg2tH,  Wg2tL,  Dh, De);
  transpose_split_kernel<<<tg2, tb, 0, stream>>>(mWg2, mWg2tH, mWg2tL, Dh, De);

  // ---- 3) layer-1 GEMMs (ReLU + split output) ----
  dim3 g1d(M / 128, Dh / 128), g1g(B / 128, Dh / 128);
  gemm_bf16x3<0><<<g1d, 256, 0, stream>>>(XqH, XqL, Wd1tH, Wd1tL, bd1, M, Dh, C, HdqH, HdqL, nullptr);
  gemm_bf16x3<0><<<g1d, 256, 0, stream>>>(XkH, XkL, mWd1tH, mWd1tL, mbd1, M, Dh, C, HdkH, HdkL, nullptr);
  gemm_bf16x3<0><<<g1g, 256, 0, stream>>>(GqH, GqL, Wg1tH, Wg1tL, bg1, B, Dh, C, HgqH, HgqL, nullptr);
  gemm_bf16x3<0><<<g1g, 256, 0, stream>>>(GkH, GkL, mWg1tH, mWg1tL, mbg1, B, Dh, C, HgkH, HgkL, nullptr);

  // ---- 4) layer-2 GEMMs (fp32 output) ----
  dim3 g2d(M / 128, De / 128), g2g(B / 128, De / 128);
  gemm_bf16x3<1><<<g2d, 256, 0, stream>>>(HdqH, HdqL, Wd2tH, Wd2tL, bd2, M, De, Dh, nullptr, nullptr, qd);
  gemm_bf16x3<1><<<g2d, 256, 0, stream>>>(HdkH, HdkL, mWd2tH, mWd2tL, mbd2, M, De, Dh, nullptr, nullptr, kd);
  gemm_bf16x3<1><<<g2g, 256, 0, stream>>>(HgqH, HgqL, Wg2tH, Wg2tL, bg2, B, De, Dh, nullptr, nullptr, qg);
  gemm_bf16x3<1><<<g2g, 256, 0, stream>>>(HgkH, HgkL, mWg2tH, mWg2tL, mbg2, B, De, Dh, nullptr, nullptr, kg);

  // ---- 5) l2 normalize ----
  l2norm_kernel<<<M / 4, 128, 0, stream>>>(qd, M, De);
  l2norm_kernel<<<M / 4, 128, 0, stream>>>(kd, M, De);
  l2norm_kernel<<<B / 4, 128, 0, stream>>>(qg, B, De);
  l2norm_kernel<<<B / 4, 128, 0, stream>>>(kg, B, De);

  // ---- 6) losses ----
  global_nce_kernel<<<B, 128, 0, stream>>>(qg, kg, pg);
  argmax_kernel<<<B, 64, 0, stream>>>(qd, kd, midx);
  dense_nce_kernel<<<M, 256, 0, stream>>>(qd, kd, midx, pd, M);
  final_kernel<<<1, 256, 0, stream>>>(pg, B, pd, M, out);

  (void)in_sizes; (void)n_in; (void)out_size; (void)ws_size;
}
